// Head_27968827031692
// MI455X (gfx1250) — matrix operand order repaired and checked by reading, not compiled, not run on hardware
//
#include <hip/hip_runtime.h>

// ---------------------------------------------------------------------------
// Causal attention head for MI455X (gfx1250): bf16 WMMA flash-attention.
//   B=16, T=2048, C=1024, H=128, fp32 in/out, bf16 tensor-core math.
//   - all LDS fragment loads are 2x ds_load_b128 (contiguous 16B runs)
//   - K/V tiles double-buffered via GLOBAL_LOAD_ASYNC_TO_LDS_B128, with
//     in-order ASYNCcnt waits (s_wait_asynccnt 4) hiding load latency
//   - softmax scale folded into the Wq projection
// ---------------------------------------------------------------------------

typedef __attribute__((ext_vector_type(16))) __bf16 v16bf;
typedef __attribute__((ext_vector_type(8)))  __bf16 v8bf;
typedef __attribute__((ext_vector_type(8)))  float  v8f;

constexpr int Bdim = 16;
constexpr int Tdim = 2048;
constexpr int Cdim = 1024;
constexpr int Hdim = 128;
constexpr int BM   = 128;   // rows per workgroup (8 waves x 16)
constexpr int BK   = 32;    // WMMA K chunk
constexpr float SM_SCALE = 0.08838834764831845f;  // 1/sqrt(128)

// float -> bf16 bits (round-half-up), storage-only
static __device__ inline __bf16 f2b(float f) {
  unsigned u = __builtin_bit_cast(unsigned, f);
  unsigned short h = (unsigned short)((u + 0x8000u) >> 16);
  return __builtin_bit_cast(__bf16, h);
}

// pack two floats into one dword of 2x bf16 (f0 -> low half)
static __device__ inline unsigned pk2(float f0, float f1) {
  unsigned u0 = __builtin_bit_cast(unsigned, f0) + 0x8000u;
  unsigned u1 = __builtin_bit_cast(unsigned, f1) + 0x8000u;
#if __has_builtin(__builtin_amdgcn_perm)
  return __builtin_amdgcn_perm(u1, u0, 0x07060302u);  // one v_perm_b32
#else
  return (u0 >> 16) | (u1 & 0xFFFF0000u);
#endif
}

// ---- CDNA5 async global->LDS (ASYNCcnt-tracked, VGPR-bypassing) -----------
static __device__ inline void async_ld128(unsigned lds_addr, unsigned voff,
                                          unsigned long long gbase) {
  asm volatile("global_load_async_to_lds_b128 %0, %1, %2"
               :: "v"(lds_addr), "v"(voff), "s"(gbase) : "memory");
}
template <int N> static __device__ inline void wait_async_le() {
  asm volatile("s_wait_asynccnt %0" :: "i"(N) : "memory");
}
static __device__ inline unsigned lds_off(const void* p) {
  return (unsigned)(uintptr_t)p;   // LDS aperture: low 32 bits = LDS offset
}

// Combine two contiguous 8-element (16B) LDS runs into a v16bf fragment.
static __device__ inline v16bf frag2(const __bf16* p0, const __bf16* p1) {
  v8bf a = *(const v8bf*)p0;
  v8bf b = *(const v8bf*)p1;
  return __builtin_shufflevector(a, b, 0, 1, 2, 3, 4, 5, 6, 7,
                                       8, 9, 10, 11, 12, 13, 14, 15);
}

// A fragment (16x32): lane = row, element i -> k = i + (i>=8?8:0) + half*8.
static __device__ inline v16bf ldsA(const __bf16* base, int row, int ld,
                                    int kbase, int half) {
  const __bf16* p = base + row * ld + kbase + half * 8;
  return frag2(p, p + 16);
}

// B fragment (32x16): lane = col, element i -> k = i + (i>=8?8:0) + half*8.
static __device__ inline v16bf ldsB(const __bf16* base, int col, int ld,
                                    int kbase, int half) {
  const __bf16* p = base + col * ld + kbase + half * 8;
  return frag2(p, p + 16);
}

// ---------------------------------------------------------------------------
// QKV projection: Out = X(f32)[M=B*T][C] * (wscale * W(f32)[C][H]) -> bf16.
// transposedOut=0: Out[m][h].   transposedOut=1: Out[b][h][t].
// ---------------------------------------------------------------------------
__global__ __launch_bounds__(256) void qkv_proj(const float* __restrict__ X,
                                                const float* __restrict__ W,
                                                __bf16* __restrict__ Out,
                                                int transposedOut,
                                                float wscale) {
  __shared__ __align__(16) __bf16 xs[BM * 40];    // x tile 128x32, ld 40
  __shared__ __align__(16) __bf16 wt[Hdim * 40];  // W tile transposed 128x32

  const int tid = threadIdx.x;
  const int lane = tid & 31, wv = tid >> 5;
  const int half = lane >> 4, lr = lane & 15;
  const int m0 = blockIdx.x * BM;

  v8f acc[8];
#pragma unroll
  for (int n = 0; n < 8; ++n)
#pragma unroll
    for (int r = 0; r < 8; ++r) acc[n][r] = 0.0f;

  for (int c0 = 0; c0 < Cdim; c0 += BK) {
    __syncthreads();
#pragma unroll
    for (int i = 0; i < 8; ++i) {          // x tile: fp32 pairs -> bf16 dword
      int idx = tid + 256 * i;
      int r = idx >> 4, j = idx & 15;
      const float* xp = &X[(size_t)(m0 + r) * Cdim + c0 + 2 * j];
      *(unsigned*)&xs[r * 40 + 2 * j] = pk2(xp[0], xp[1]);
    }
#pragma unroll
    for (int i = 0; i < 8; ++i) {          // W tile, transposed + scaled
      int idx = tid + 256 * i;
      int h = idx & 127, j = idx >> 7;
      float f0 = W[(size_t)(c0 + 2 * j) * Hdim + h] * wscale;
      float f1 = W[(size_t)(c0 + 2 * j + 1) * Hdim + h] * wscale;
      *(unsigned*)&wt[h * 40 + 2 * j] = pk2(f0, f1);
    }
    __syncthreads();

    v16bf a = ldsA(xs, wv * 16 + lr, 40, 0, half);
#pragma unroll
    for (int n = 0; n < 8; ++n) {
      v16bf b = ldsB(wt, n * 16 + lr, 40, 0, half);
      acc[n] = __builtin_amdgcn_wmma_f32_16x16x32_bf16(
          false, a, false, b, (short)0, acc[n], false, false);
    }
  }

  if (!transposedOut) {
#pragma unroll
    for (int n = 0; n < 8; ++n)
#pragma unroll
      for (int r = 0; r < 8; ++r) {
        int m = m0 + wv * 16 + r + 8 * half;
        Out[(size_t)m * Hdim + n * 16 + lr] = f2b(acc[n][r]);
      }
  } else {
    // C-layout rows for fixed n are CONSECUTIVE t at fixed h -> one 16B store
    const int bb = m0 / Tdim;
    const int t0 = m0 - bb * Tdim;
    const size_t ob = (size_t)bb * Hdim * Tdim;
    const int t = t0 + wv * 16 + 8 * half;
#pragma unroll
    for (int n = 0; n < 8; ++n) {
      uint4 v;
      v.x = pk2(acc[n][0], acc[n][1]);
      v.y = pk2(acc[n][2], acc[n][3]);
      v.z = pk2(acc[n][4], acc[n][5]);
      v.w = pk2(acc[n][6], acc[n][7]);
      *(uint4*)&Out[ob + (size_t)(n * 16 + lr) * Tdim + t] = v;
    }
  }
}

// ---------------------------------------------------------------------------
// Flash attention, causal.  Q/K in [B][T][H] bf16 (scale pre-folded into Q),
// V in [B][H][T] bf16.  Double-buffered async K/V tiles.
// ---------------------------------------------------------------------------
__global__ __launch_bounds__(256) void flash_attn(const __bf16* __restrict__ Q,
                                                  const __bf16* __restrict__ K,
                                                  const __bf16* __restrict__ Vt,
                                                  float* __restrict__ Out) {
  // Layout (elements):  Ks0 [0,4352)  Ks1 [4352,8704)  Vs0 [8704,13824)
  //                     Vs1 [13824,18944)  Ps [18944,24064)
  // Q staging (17408) aliases [4352, 21760) -- disjoint from Ks0.
  __shared__ __align__(16) __bf16 smem[24064];            // 48.1 KB
  __bf16* Ks0 = smem;
  __bf16* Ks1 = smem + 4352;
  __bf16* Vs0 = smem + 8704;
  __bf16* Vs1 = smem + 13824;
  __bf16* Ps  = smem + 18944;      // 8 waves x 16 x 40
  __bf16* Qst = smem + 4352;       // Q staging alias, ld 136

  const int tid = threadIdx.x;
  const int lane = tid & 31, wv = tid >> 5;
  const int half = lane >> 4, lr = lane & 15;
  const int b  = blockIdx.y;
  const int q0 = blockIdx.x * BM;
  const size_t base  = (size_t)b * Tdim * Hdim;
  const size_t vbase = (size_t)b * Hdim * Tdim;

  // Per-thread loop-invariant tile offsets.
  const int kr = tid >> 4, kc = tid & 15;        // K tile: 32 x 16 chunks
  const unsigned voffK0 = (unsigned)((kr * Hdim + kc * 8) * 2);
  const unsigned voffK1 = (unsigned)(((kr + 16) * Hdim + kc * 8) * 2);
  const unsigned ldsK0b[2] = { lds_off(&Ks0[kr * 136 + kc * 8]),
                               lds_off(&Ks1[kr * 136 + kc * 8]) };
  const unsigned ldsK1b[2] = { lds_off(&Ks0[(kr + 16) * 136 + kc * 8]),
                               lds_off(&Ks1[(kr + 16) * 136 + kc * 8]) };
  const int vh = tid >> 2, vc = tid & 3;         // V tile: 128 x 4 chunks
  const unsigned voffV0 = (unsigned)((vh * Tdim + vc * 8) * 2);
  const unsigned voffV1 = (unsigned)(((vh + 64) * Tdim + vc * 8) * 2);
  const unsigned ldsV0b[2] = { lds_off(&Vs0[vh * 40 + vc * 8]),
                               lds_off(&Vs1[vh * 40 + vc * 8]) };
  const unsigned ldsV1b[2] = { lds_off(&Vs0[(vh + 64) * 40 + vc * 8]),
                               lds_off(&Vs1[(vh + 64) * 40 + vc * 8]) };

  // ---- prologue: async-stage Q (aliased region) and K tile 0 (Ks0) ----
  {
    const unsigned long long gQ =
        (unsigned long long)(uintptr_t)(Q + base + (size_t)q0 * Hdim);
#pragma unroll
    for (int i = 0; i < 8; ++i) {
      int idx = tid + 256 * i;
      int r = idx >> 4, c = idx & 15;
      async_ld128(lds_off(&Qst[r * 136 + c * 8]),
                  (unsigned)((r * Hdim + c * 8) * 2), gQ);
    }
    const unsigned long long gK0 = (unsigned long long)(uintptr_t)(K + base);
    async_ld128(ldsK0b[0], voffK0, gK0);
    async_ld128(ldsK1b[0], voffK1, gK0);
  }
  wait_async_le<2>();        // in-order: the 8 Q chunks have landed
  __syncthreads();
  v16bf qf[4];
#pragma unroll
  for (int kk = 0; kk < 4; ++kk)
    qf[kk] = ldsA(Qst, wv * 16 + lr, 136, kk * 32, half);
  __syncthreads();           // all waves done reading the aliased region
  {
    const unsigned long long gV0 = (unsigned long long)(uintptr_t)(Vt + vbase);
    async_ld128(ldsV0b[0], voffV0, gV0);
    async_ld128(ldsV1b[0], voffV1, gV0);
  }

  v8f   o[8];
  float mrow[8], lrow[8];
#pragma unroll
  for (int n = 0; n < 8; ++n)
#pragma unroll
    for (int r = 0; r < 8; ++r) o[n][r] = 0.0f;
#pragma unroll
  for (int r = 0; r < 8; ++r) { mrow[r] = -1.0e30f; lrow[r] = 0.0f; }

  const int qmaxw = q0 + wv * 16 + 15;
  const int nkb   = q0 / 32 + 4;

  for (int j = 0; j < nkb; ++j) {
    const int k0  = j * 32;
    const int cur = j & 1, nxt = cur ^ 1;
    if (j > 0) __syncthreads();          // readers of buf[nxt] (iter j-1) done
    const bool pre = (j + 1 < nkb);
    if (pre) {                           // prefetch next tile into buf[nxt]
      const unsigned long long gK = (unsigned long long)(uintptr_t)(
          K + base + (size_t)(k0 + 32) * Hdim);
      const unsigned long long gV = (unsigned long long)(uintptr_t)(
          Vt + vbase + k0 + 32);
      async_ld128(ldsK0b[nxt], voffK0, gK);
      async_ld128(ldsK1b[nxt], voffK1, gK);
      async_ld128(ldsV0b[nxt], voffV0, gV);
      async_ld128(ldsV1b[nxt], voffV1, gV);
      wait_async_le<4>();   // in-order: all but the 4 just issued are done
    } else {
      wait_async_le<0>();
    }
    __syncthreads();                     // current tile visible to all waves

    const __bf16* Kc = cur ? Ks1 : Ks0;
    const __bf16* Vc = cur ? Vs1 : Vs0;

    if (k0 <= qmaxw) {                   // wave-uniform; EXEC stays full
      // ---- S = Q * K^T (16 x 32); scale pre-folded into Q ----
      v8f s0, s1;
#pragma unroll
      for (int r = 0; r < 8; ++r) { s0[r] = 0.0f; s1[r] = 0.0f; }
#pragma unroll
      for (int kk = 0; kk < 4; ++kk) {
        v16bf b0 = ldsB(Kc, lr,      136, kk * 32, half);
        v16bf b1 = ldsB(Kc, 16 + lr, 136, kk * 32, half);
        s0 = __builtin_amdgcn_wmma_f32_16x16x32_bf16(
            false, qf[kk], false, b0, (short)0, s0, false, false);
        s1 = __builtin_amdgcn_wmma_f32_16x16x32_bf16(
            false, qf[kk], false, b1, (short)0, s1, false, false);
      }

      // ---- online softmax (C-layout: row = r+8*half, col = lr) ----
      const int qrb = q0 + wv * 16 + 8 * half;
      float sc0[8], sc1[8], bm[8];
#pragma unroll
      for (int r = 0; r < 8; ++r) {
        int qr = qrb + r;
        sc0[r] = (k0 + lr      > qr) ? -1.0e30f : s0[r];
        sc1[r] = (k0 + 16 + lr > qr) ? -1.0e30f : s1[r];
        bm[r]  = fmaxf(sc0[r], sc1[r]);
      }
#pragma unroll
      for (int off = 1; off < 16; off <<= 1)
#pragma unroll
        for (int r = 0; r < 8; ++r)
          bm[r] = fmaxf(bm[r], __shfl_xor(bm[r], off, 32));

      float p0[8], p1[8], bs[8], alpha[8];
#pragma unroll
      for (int r = 0; r < 8; ++r) {
        float mn = fmaxf(mrow[r], bm[r]);
        alpha[r] = __expf(mrow[r] - mn);
        mrow[r]  = mn;
        p0[r] = __expf(sc0[r] - mn);
        p1[r] = __expf(sc1[r] - mn);
        bs[r] = p0[r] + p1[r];
      }
#pragma unroll
      for (int off = 1; off < 16; off <<= 1)
#pragma unroll
        for (int r = 0; r < 8; ++r) bs[r] += __shfl_xor(bs[r], off, 32);
#pragma unroll
      for (int r = 0; r < 8; ++r) lrow[r] = lrow[r] * alpha[r] + bs[r];
#pragma unroll
      for (int n = 0; n < 8; ++n)
#pragma unroll
        for (int r = 0; r < 8; ++r) o[n][r] *= alpha[r];

      // ---- P: C-layout -> A-fragment via per-wave LDS round trip ----
      __bf16* Pw = Ps + wv * (16 * 40);
#pragma unroll
      for (int r = 0; r < 8; ++r) {
        int m = r + 8 * half;
        Pw[m * 40 + lr]      = f2b(p0[r]);
        Pw[m * 40 + 16 + lr] = f2b(p1[r]);
      }
      asm volatile("s_wait_dscnt 0" ::: "memory");  // wave-local RAW fence
      v16bf pf = ldsA(Pw, lr, 40, 0, half);

      // ---- O += P * V (16 x 128); transposed V -> contiguous B runs ----
#pragma unroll
      for (int n = 0; n < 8; ++n) {
        v16bf vb = ldsB(Vc, n * 16 + lr, 40, 0, half);
        o[n] = __builtin_amdgcn_wmma_f32_16x16x32_bf16(
            false, pf, false, vb, (short)0, o[n], false, false);
      }
    }
  }

  // ---- normalize and store fp32 ----
#pragma unroll
  for (int n = 0; n < 8; ++n)
#pragma unroll
    for (int r = 0; r < 8; ++r) {
      int qr = q0 + wv * 16 + r + 8 * half;
      Out[base + (size_t)qr * Hdim + n * 16 + lr] = o[n][r] / lrow[r];
    }
}

// ---------------------------------------------------------------------------
extern "C" void kernel_launch(void* const* d_in, const int* in_sizes, int n_in,
                              void* d_out, int out_size, void* d_ws, size_t ws_size,
                              hipStream_t stream) {
  const float* x  = (const float*)d_in[0];
  const float* Wk = (const float*)d_in[1];
  const float* Wq = (const float*)d_in[2];
  const float* Wv = (const float*)d_in[3];
  float* out = (float*)d_out;

  const size_t MH = (size_t)Bdim * Tdim * Hdim;
  __bf16* qb = (__bf16*)d_ws;                     // [B][T][H], scale folded
  __bf16* kb = qb + MH;                           // [B][T][H]
  __bf16* vb = kb + MH;                           // [B][H][T] (transposed)

  dim3 gp((Bdim * Tdim) / BM);
  qkv_proj<<<gp, 256, 0, stream>>>(x, Wq, qb, 0, SM_SCALE);
  qkv_proj<<<gp, 256, 0, stream>>>(x, Wk, kb, 0, 1.0f);
  qkv_proj<<<gp, 256, 0, stream>>>(x, Wv, vb, 1, 1.0f);

  dim3 ga(Tdim / BM, Bdim);
  flash_attn<<<ga, 256, 0, stream>>>(qb, kb, vb, out);
}
